// BASE_2525440770953
// MI455X (gfx1250) — matrix operand order repaired and checked by reading, not compiled, not run on hardware
//
#include <hip/hip_runtime.h>
#include <math.h>

typedef __attribute__((ext_vector_type(16))) _Float16 v16h;
typedef __attribute__((ext_vector_type(8)))  _Float16 v8h;
typedef __attribute__((ext_vector_type(8)))  float    v8f;

// ======================= generic f16 WMMA GEMM (B transposed) =======================
// C[M][N] = A[M][K] * B[K][N] with Bt = B^T given as [N][K] row-major.
// fp32 accumulate. M,N mult of 16, K mult of 64. One wave32 per 16x16 C tile.
// Both fragments load as 2x contiguous v8h (global_load_b128) per operand per K-step.
__global__ __launch_bounds__(256) void wmma_gemm_f16t(
    const _Float16* __restrict__ A,   // M x K
    const _Float16* __restrict__ Bt,  // N x K
    float* __restrict__ C, int M, int N, int K)
{
  const int lane   = threadIdx.x & 31;
  const int wave   = threadIdx.x >> 5;
  const int tilesN = N >> 4;
  const int nTiles = (M >> 4) * tilesN;
  const int tile   = blockIdx.x * 8 + wave;
  if (tile >= nTiles) return;                 // wave-uniform: EXEC stays all-ones
  const int tm    = (tile / tilesN) << 4;
  const int tn    = (tile % tilesN) << 4;
  const int idx16 = lane & 15;                // row-in-tile (A) / col-in-tile (B,C)
  const int half  = lane >> 4;                // K/M half selector (ISA 7.12.2)

  const _Float16* arow = A  + (size_t)(tm + idx16) * K;
  const _Float16* brow = Bt + (size_t)(tn + idx16) * K;
  v8f acc0 = {}, acc1 = {};
  for (int k0 = 0; k0 < K; k0 += 64) {
    // ---- K sub-step 0 ----
    v8h alo = *(const v8h*)(arow + k0 + half * 8);
    v8h ahi = *(const v8h*)(arow + k0 + 16 + half * 8);
    v8h blo = *(const v8h*)(brow + k0 + half * 8);
    v8h bhi = *(const v8h*)(brow + k0 + 16 + half * 8);
    v16h a, b;
#pragma unroll
    for (int i = 0; i < 8; ++i) { a[i] = alo[i]; a[8+i] = ahi[i];
                                  b[i] = blo[i]; b[8+i] = bhi[i]; }
    acc0 = __builtin_amdgcn_wmma_f32_16x16x32_f16(
        false, a, false, b, (short)0, acc0, false, false);
    // ---- K sub-step 1 (independent accumulator hides WMMA RAW latency) ----
    v8h alo1 = *(const v8h*)(arow + k0 + 32 + half * 8);
    v8h ahi1 = *(const v8h*)(arow + k0 + 48 + half * 8);
    v8h blo1 = *(const v8h*)(brow + k0 + 32 + half * 8);
    v8h bhi1 = *(const v8h*)(brow + k0 + 48 + half * 8);
    v16h a1, b1;
#pragma unroll
    for (int i = 0; i < 8; ++i) { a1[i] = alo1[i]; a1[8+i] = ahi1[i];
                                  b1[i] = blo1[i]; b1[8+i] = bhi1[i]; }
    acc1 = __builtin_amdgcn_wmma_f32_16x16x32_f16(
        false, a1, false, b1, (short)0, acc1, false, false);
    __builtin_prefetch(arow + k0 + 128, 0, 1);   // -> global_prefetch_b8
    __builtin_prefetch(brow + k0 + 128, 0, 1);
  }
#pragma unroll
  for (int r = 0; r < 8; ++r)
    C[(size_t)(tm + half * 8 + r) * N + tn + idx16] = acc0[r] + acc1[r];
}

// ======================= SKConv stage =======================
__global__ __launch_bounds__(256) void sk_grouped_conv(
    const float* __restrict__ x, const float* __restrict__ w,
    const float* __restrict__ bias, float* __restrict__ out, int ksz, int pad)
{
  const int oc = blockIdx.y;
  const int p  = blockIdx.x * 256 + threadIdx.x;      // 0..1023
  const int y  = p >> 5, xq = p & 31;
  const int icbase = (oc >> 4) << 4;                   // group * 16
  const float* wp = w + (size_t)oc * 16 * ksz * ksz;
  float acc = bias[oc];
  for (int ic = 0; ic < 16; ++ic) {
    const float* xp = x + (size_t)(icbase + ic) * 1024;
    const float* wr = wp + ic * ksz * ksz;
    for (int ky = 0; ky < ksz; ++ky) {
      const int yy = y + ky - pad;
      if ((unsigned)yy >= 32u) continue;
      for (int kx = 0; kx < ksz; ++kx) {
        const int xx = xq + kx - pad;
        if ((unsigned)xx >= 32u) continue;
        acc = fmaf(xp[yy * 32 + xx], wr[ky * ksz + kx], acc);
      }
    }
  }
  out[(size_t)oc * 1024 + p] = acc;
}

// per-plane (1024 px) instance norm + activation. act: 0=relu 1=leaky(0.2) 2=none
__global__ __launch_bounds__(256) void instnorm_act(
    const float* __restrict__ in, float* __restrict__ out, int act)
{
  const float* p = in + (size_t)blockIdx.x * 1024;
  float* o = out + (size_t)blockIdx.x * 1024;
  const int tid = threadIdx.x;
  float v0 = p[tid], v1 = p[tid + 256], v2 = p[tid + 512], v3 = p[tid + 768];
  float s  = v0 + v1 + v2 + v3;
  float ss = v0 * v0 + v1 * v1 + v2 * v2 + v3 * v3;
  for (int off = 16; off > 0; off >>= 1) {
    s  += __shfl_down(s,  off, 32);
    ss += __shfl_down(ss, off, 32);
  }
  __shared__ float sh_s[8], sh_q[8];
  const int w = tid >> 5, l = tid & 31;
  if (l == 0) { sh_s[w] = s; sh_q[w] = ss; }
  __syncthreads();
  if (tid == 0) {
    float ts = 0.f, tq = 0.f;
    for (int i = 0; i < 8; ++i) { ts += sh_s[i]; tq += sh_q[i]; }
    const float mu = ts * (1.f / 1024.f);
    sh_s[0] = mu;
    sh_q[0] = tq * (1.f / 1024.f) - mu * mu;
  }
  __syncthreads();
  const float mu   = sh_s[0];
  const float rstd = rsqrtf(sh_q[0] + 1e-5f);
  float r[4] = { v0, v1, v2, v3 };
#pragma unroll
  for (int i = 0; i < 4; ++i) {
    float v = (r[i] - mu) * rstd;
    if (act == 0)      v = fmaxf(v, 0.f);
    else if (act == 1) v = (v >= 0.f) ? v : 0.2f * v;
    o[tid + i * 256] = v;
  }
}

// fea_s[c] = (sum over 3 branches) mean over pixels
__global__ __launch_bounds__(256) void feas_pool(
    const float* __restrict__ feas, float* __restrict__ fea_s)
{
  const int c = blockIdx.x;
  float s = 0.f;
  for (int b = 0; b < 3; ++b) {
    const float* p = feas + ((size_t)b * 512 + c) * 1024;
    for (int i = threadIdx.x; i < 1024; i += 256) s += p[i];
  }
  __shared__ float red[256];
  red[threadIdx.x] = s; __syncthreads();
  for (int st = 128; st > 0; st >>= 1) {
    if (threadIdx.x < st) red[threadIdx.x] += red[threadIdx.x + st];
    __syncthreads();
  }
  if (threadIdx.x == 0) fea_s[c] = red[0] * (1.f / 1024.f);
}

// fc -> 3 branch logits -> softmax over branches. One block, 512 threads.
__global__ __launch_bounds__(512) void sk_attention(
    const float* __restrict__ fea_s,
    const float* __restrict__ w_fc,  const float* __restrict__ b_fc,
    const float* __restrict__ w0, const float* __restrict__ b0,
    const float* __restrict__ w1, const float* __restrict__ b1,
    const float* __restrict__ w2, const float* __restrict__ b2,
    float* __restrict__ att)
{
  __shared__ float fs[512];
  __shared__ float z[32];
  fs[threadIdx.x] = fea_s[threadIdx.x];
  __syncthreads();
  if (threadIdx.x < 32) {
    float a = b_fc[threadIdx.x];
    const float* wr = w_fc + threadIdx.x * 512;
    for (int c = 0; c < 512; ++c) a = fmaf(fs[c], wr[c], a);
    z[threadIdx.x] = a;
  }
  __syncthreads();
  const int c = threadIdx.x;
  float l0 = b0[c], l1 = b1[c], l2 = b2[c];
  for (int j = 0; j < 32; ++j) {
    const float zj = z[j];
    l0 = fmaf(zj, w0[c * 32 + j], l0);
    l1 = fmaf(zj, w1[c * 32 + j], l1);
    l2 = fmaf(zj, w2[c * 32 + j], l2);
  }
  const float m = fmaxf(l0, fmaxf(l1, l2));
  const float e0 = expf(l0 - m), e1 = expf(l1 - m), e2 = expf(l2 - m);
  const float inv = 1.f / (e0 + e1 + e2);
  att[c] = e0 * inv; att[512 + c] = e1 * inv; att[1024 + c] = e2 * inv;
}

__global__ void sk_combine(const float* __restrict__ feas,
                           const float* __restrict__ att,
                           float* __restrict__ out32)
{
  const int idx = blockIdx.x * 256 + threadIdx.x;   // 512*1024
  const int c = idx >> 10;
  out32[idx] = att[c] * feas[idx]
             + att[512 + c]  * feas[524288 + idx]
             + att[1024 + c] * feas[1048576 + idx];
}

// ======================= RAL stage =======================
__global__ void avgpool2(const float* __restrict__ out32, float* __restrict__ fg)
{
  const int idx = blockIdx.x * 256 + threadIdx.x;   // 512*256
  const int c = idx >> 8, q = idx & 255;
  const int y = q >> 4, x = q & 15;
  const float* pl = out32 + (size_t)c * 1024 + (y * 2) * 32 + x * 2;
  fg[idx] = 0.25f * (pl[0] + pl[1] + pl[32] + pl[33]);
}

__global__ __launch_bounds__(256) void ral_invnorm(
    const float* __restrict__ fg, float* __restrict__ invnorm)
{
  const int q = blockIdx.x, qy = q >> 4, qx = q & 15;
  float s = 0.f;
  for (int c = threadIdx.x; c < 512; c += 256) {
    const float* pl = fg + (size_t)c * 256;
#pragma unroll
    for (int u = 0; u < 3; ++u) {
      const int yy = qy + u - 1;
      if ((unsigned)yy >= 16u) continue;
#pragma unroll
      for (int v = 0; v < 3; ++v) {
        const int xx = qx + v - 1;
        if ((unsigned)xx >= 16u) continue;
        const float t = pl[yy * 16 + xx];
        s += t * t;
      }
    }
  }
  __shared__ float red[256];
  red[threadIdx.x] = s; __syncthreads();
  for (int st = 128; st > 0; st >>= 1) {
    if (threadIdx.x < st) red[threadIdx.x] += red[threadIdx.x + st];
    __syncthreads();
  }
  if (threadIdx.x == 0) invnorm[q] = 1.f / fmaxf(sqrtf(red[0]), 1e-4f);
}

// im2col^T of padded fg_small: colT[p][k=(c*9+u*3+v)] (f16), coalesced writes.
// Serves as BOTH GEMM operands of the symmetric score Gram matrix.
__global__ void ral_im2col(const float* __restrict__ fg, _Float16* __restrict__ colT)
{
  const int idx = blockIdx.x * 256 + threadIdx.x;   // 256*4608
  const int p = idx / 4608, k = idx % 4608;
  const int c = k / 9, t = k % 9, u = t / 3, v = t % 3;
  const int yy = (p >> 4) + u - 1, xx = (p & 15) + v - 1;
  float val = 0.f;
  if ((unsigned)yy < 16u && (unsigned)xx < 16u) val = fg[(size_t)c * 256 + yy * 16 + xx];
  colT[idx] = (_Float16)val;
}

// softmax over q (256) per coarse pixel p, scaled by 10*invnorm[q]
__global__ __launch_bounds__(256) void ral_softmax(
    const float* __restrict__ raw, const float* __restrict__ invnorm,
    float* __restrict__ attn)
{
  const int p = blockIdx.x, q = threadIdx.x;
  const float v = raw[(size_t)q * 256 + p] * 10.f * invnorm[q];
  __shared__ float red[256];
  red[q] = v; __syncthreads();
  for (int st = 128; st > 0; st >>= 1) {
    if (q < st) red[q] = fmaxf(red[q], red[q + st]);
    __syncthreads();
  }
  const float m = red[0]; __syncthreads();
  const float e = expf(v - m);
  red[q] = e; __syncthreads();
  for (int st = 128; st > 0; st >>= 1) {
    if (q < st) red[q] += red[q + st];
    __syncthreads();
  }
  attn[(size_t)q * 256 + p] = e / red[0];
}

// conv_transpose2d(attn, bg_patches, stride=2, pad=1)/4; patches read from out32 on the fly
__global__ __launch_bounds__(256) void ral_deconv(
    const float* __restrict__ attn, const float* __restrict__ out32,
    float* __restrict__ out_ral)
{
  const int c = blockIdx.y;
  const int pix = blockIdx.x * 256 + threadIdx.x;
  const int Y = pix >> 5, X = pix & 31;
  const float* plane = out32 + (size_t)c * 1024;
  float acc = 0.f;
  for (int ky = 0; ky < 4; ++ky) {
    const int yn = Y + 1 - ky;
    if (yn < 0 || (yn & 1)) continue;
    const int y = yn >> 1; if (y > 15) continue;
    for (int kx = 0; kx < 4; ++kx) {
      const int xn = X + 1 - kx;
      if (xn < 0 || (xn & 1)) continue;
      const int x = xn >> 1; if (x > 15) continue;
      const float* acol = attn + (y * 16 + x);
      for (int qy = 0; qy < 16; ++qy) {
        const int ry = 2 * qy + ky - 1;
        if ((unsigned)ry >= 32u) continue;
        const float* prow = plane + ry * 32;
        for (int qx = 0; qx < 16; ++qx) {
          const int rx = 2 * qx + kx - 1;
          if ((unsigned)rx >= 32u) continue;
          acc = fmaf(acol[(size_t)(qy * 16 + qx) * 256], prow[rx], acc);
        }
      }
    }
  }
  out_ral[(size_t)c * 1024 + pix] = 0.25f * acc;
}

// ======================= CSA stage =======================
__global__ void sigmoid_k(const float* __restrict__ in, float* __restrict__ out, int n)
{
  const int i = blockIdx.x * 256 + threadIdx.x;
  if (i < n) out[i] = 1.f / (1.f + expf(-in[i]));
}

__global__ __launch_bounds__(256) void csa_weights(
    const float* __restrict__ sig, float* __restrict__ csa_w)
{
  const int p = blockIdx.x, y = p >> 5, x = p & 31, tid = threadIdx.x;
  float acc[9];
#pragma unroll
  for (int t = 0; t < 9; ++t) acc[t] = 0.f;
  for (int c = tid; c < 512; c += 256) {
    const float* base = sig + (size_t)c * 1024;
    const float pc = base[p];
    int t = 0;
#pragma unroll
    for (int u = 0; u < 3; ++u) {
      const int yy = y + u - 1;
#pragma unroll
      for (int v = 0; v < 3; ++v, ++t) {
        const int xx = x + v - 1;
        if ((unsigned)yy < 32u && (unsigned)xx < 32u)
          acc[t] = fmaf(pc, base[yy * 32 + xx], acc[t]);
      }
    }
  }
  __shared__ float red[256];
  __shared__ float res[9];
#pragma unroll
  for (int t = 0; t < 9; ++t) {
    red[tid] = acc[t]; __syncthreads();
    for (int st = 128; st > 0; st >>= 1) {
      if (tid < st) red[tid] += red[tid + st];
      __syncthreads();
    }
    if (tid == 0) res[t] = red[0] * (1.f / 512.f);
    __syncthreads();
  }
  if (tid == 0) {
    float m = res[0];
#pragma unroll
    for (int t = 1; t < 9; ++t) m = fmaxf(m, res[t]);
    float e[9]; float sum = 0.f;
#pragma unroll
    for (int t = 0; t < 9; ++t) { e[t] = expf(res[t] - m); sum += e[t]; }
    const float inv = 1.f / sum;
#pragma unroll
    for (int t = 0; t < 9; ++t) csa_w[p * 9 + t] = e[t] * inv;
  }
}

// out_csa stored p-major: flat[p*512+c]  (reference raw-reshape)
__global__ void csa_apply(const float* __restrict__ csa_w,
                          const float* __restrict__ out_ral,
                          float* __restrict__ out_csa)
{
  const int idx = blockIdx.x * 256 + threadIdx.x;   // 512*1024
  const int c = idx >> 10, p = idx & 1023;
  const int y = p >> 5, x = p & 31;
  const float* plane = out_ral + (size_t)c * 1024;
  const float* w = csa_w + p * 9;
  float acc = 0.f; int t = 0;
#pragma unroll
  for (int u = 0; u < 3; ++u) {
    const int yy = y + u - 1;
#pragma unroll
    for (int v = 0; v < 3; ++v, ++t) {
      const int xx = x + v - 1;
      if ((unsigned)yy < 32u && (unsigned)xx < 32u)
        acc = fmaf(w[t], plane[yy * 32 + xx], acc);
    }
  }
  out_csa[(size_t)p * 512 + c] = acc;
}

// ======================= layout / conversion helpers =======================
__global__ void f32_to_f16(const float* __restrict__ in, _Float16* __restrict__ out, int n)
{
  const int i = blockIdx.x * 256 + threadIdx.x;
  if (i < n) out[i] = (_Float16)in[i];
}

// z^T = concat(a[512 ch x 1024 px], b[512 ch x 1024 px]) transposed -> f16 [1024 px][1024 ch]
__global__ void concat_t_f16(const float* __restrict__ a, const float* __restrict__ b,
                             _Float16* __restrict__ out)
{
  const int idx = blockIdx.x * 256 + threadIdx.x;   // 1024*1024, out index [p][i]
  const int p = idx >> 10, i = idx & 1023;
  const float v = (i < 512) ? a[(size_t)i * 1024 + p]
                            : b[(size_t)(i - 512) * 1024 + p];
  out[idx] = (_Float16)v;
}

// ======================= host orchestration =======================
extern "C" void kernel_launch(void* const* d_in, const int* in_sizes, int n_in,
                              void* d_out, int out_size, void* d_ws, size_t ws_size,
                              hipStream_t stream)
{
  const float* x      = (const float*)d_in[0];
  const float* gus    = (const float*)d_in[1];
  const float* w_sk3  = (const float*)d_in[2];
  const float* b_sk3  = (const float*)d_in[3];
  const float* w_sk5  = (const float*)d_in[4];
  const float* b_sk5  = (const float*)d_in[5];
  const float* w_sk7  = (const float*)d_in[6];
  const float* b_sk7  = (const float*)d_in[7];
  const float* w_fc   = (const float*)d_in[8];
  const float* b_fc   = (const float*)d_in[9];
  const float* w_fc0  = (const float*)d_in[10];
  const float* b_fc0  = (const float*)d_in[11];
  const float* w_fc1  = (const float*)d_in[12];
  const float* b_fc1  = (const float*)d_in[13];
  const float* w_fc2  = (const float*)d_in[14];
  const float* b_fc2  = (const float*)d_in[15];
  const float* w_down = (const float*)d_in[16];
  const float* w_fuse = (const float*)d_in[17];
  float* out = (float*)d_out;

  char* ws = (char*)d_ws;
  size_t off = 0;
  auto carve = [&](size_t bytes) -> void* {
    void* p = ws + off;
    off = (off + bytes + 255) & ~(size_t)255;
    return p;
  };
  float*    feas     = (float*)   carve(3 * 512 * 1024 * 4);   // conv -> IN+relu in place
  float*    fea_s    = (float*)   carve(512 * 4);
  float*    att      = (float*)   carve(1536 * 4);
  float*    out32    = (float*)   carve(512 * 1024 * 4);       // out_res
  float*    fg       = (float*)   carve(512 * 256 * 4);
  float*    invn     = (float*)   carve(256 * 4);
  _Float16* colT     = (_Float16*)carve(256 * 4608 * 2);       // A and Bt of score Gram
  float*    scoreRaw = (float*)   carve(256 * 256 * 4);
  float*    attn     = (float*)   carve(256 * 256 * 4);
  float*    out_ral  = (float*)   carve(512 * 1024 * 4);
  float*    sig      = (float*)   carve(512 * 1024 * 4);
  _Float16* gus_h    = (_Float16*)carve(1024 * 1024 * 2);
  _Float16* outral_h = (_Float16*)carve(512 * 1024 * 2);       // Bt of gus GEMM (no transpose!)
  float*    gus_out  = (float*)   carve(1024 * 512 * 4);
  float*    csa_w    = (float*)   carve(1024 * 9 * 4);
  float*    out_csa  = (float*)   carve(1024 * 512 * 4);
  _Float16* z1t_h    = (_Float16*)carve(1024 * 1024 * 2);      // [px][ch]
  _Float16* wdown_h  = (_Float16*)carve(512 * 1024 * 2);
  _Float16* wfuse_h  = (_Float16*)carve(512 * 1024 * 2);
  float*    down_t   = (float*)   carve(512 * 1024 * 4);
  _Float16* z2t_h    = (_Float16*)carve(1024 * 1024 * 2);      // [px][ch]
  float*    fuse_t   = (float*)   carve(512 * 1024 * 4);
  (void)ws_size; (void)n_in; (void)in_sizes; (void)out_size;

  // ---- SKConv: 3 grouped convs + bias, then per-plane IN + relu (in place) ----
  sk_grouped_conv<<<dim3(4, 512), 256, 0, stream>>>(x, w_sk3, b_sk3, feas,           3, 1);
  sk_grouped_conv<<<dim3(4, 512), 256, 0, stream>>>(x, w_sk5, b_sk5, feas + 524288,  5, 2);
  sk_grouped_conv<<<dim3(4, 512), 256, 0, stream>>>(x, w_sk7, b_sk7, feas + 1048576, 7, 3);
  instnorm_act<<<1536, 256, 0, stream>>>(feas, feas, /*relu*/0);
  feas_pool<<<512, 256, 0, stream>>>(feas, fea_s);
  sk_attention<<<1, 512, 0, stream>>>(fea_s, w_fc, b_fc, w_fc0, b_fc0, w_fc1, b_fc1,
                                      w_fc2, b_fc2, att);
  sk_combine<<<2048, 256, 0, stream>>>(feas, att, out32);

  // ---- RAL ----
  avgpool2<<<512, 256, 0, stream>>>(out32, fg);
  ral_invnorm<<<256, 256, 0, stream>>>(fg, invn);
  ral_im2col<<<4608, 256, 0, stream>>>(fg, colT);
  // score[q][p] = colT * colT^T : M=256, N=256, K=4608 (WMMA, symmetric operands)
  wmma_gemm_f16t<<<32, 256, 0, stream>>>(colT, colT, scoreRaw, 256, 256, 4608);
  ral_softmax<<<256, 256, 0, stream>>>(scoreRaw, invn, attn);
  ral_deconv<<<dim3(4, 512), 256, 0, stream>>>(attn, out32, out_ral);

  // ---- gaussian-weighted broadcast sum: gus_out[p][c] (WMMA) ----
  f32_to_f16<<<4096, 256, 0, stream>>>(gus, gus_h, 1024 * 1024);
  f32_to_f16<<<2048, 256, 0, stream>>>(out_ral, outral_h, 512 * 1024);
  // C[p][c] = sum_k gus[p][k] * out_ral[c][k]  (Bt layout is out_ral's natural layout)
  wmma_gemm_f16t<<<256, 256, 0, stream>>>(gus_h, outral_h, gus_out, 1024, 512, 1024);

  // ---- CSA ----
  sigmoid_k<<<2048, 256, 0, stream>>>(out_ral, sig, 512 * 1024);
  csa_weights<<<1024, 256, 0, stream>>>(sig, csa_w);
  csa_apply<<<2048, 256, 0, stream>>>(csa_w, out_ral, out_csa);

  // ---- fuse path ----
  // z1^T = concat(raw-reshaped gus_out, raw-reshaped out_csa)^T as [1024 px][1024 ch] f16
  concat_t_f16<<<4096, 256, 0, stream>>>(gus_out, out_csa, z1t_h);
  f32_to_f16<<<2048, 256, 0, stream>>>(w_down, wdown_h, 512 * 1024);
  f32_to_f16<<<2048, 256, 0, stream>>>(w_fuse, wfuse_h, 512 * 1024);
  // down 1x1 conv: C[o][px], A = W[o][i], Bt = z1^T[px][i]  (WMMA), then IN + leaky
  wmma_gemm_f16t<<<256, 256, 0, stream>>>(wdown_h, z1t_h, down_t, 512, 1024, 1024);
  instnorm_act<<<512, 256, 0, stream>>>(down_t, down_t, /*leaky*/1);
  // z2^T = concat(down, out_res)^T
  concat_t_f16<<<4096, 256, 0, stream>>>(down_t, out32, z2t_h);
  // fuse 1x1 conv (WMMA) + IN + leaky -> d_out
  wmma_gemm_f16t<<<256, 256, 0, stream>>>(wfuse_h, z2t_h, fuse_t, 512, 1024, 1024);
  instnorm_act<<<512, 256, 0, stream>>>(fuse_t, out, /*leaky*/1);
}
